// LSTMLayer_63290638073856
// MI455X (gfx1250) — matrix operand order repaired and checked by reading, compiled, not run on hardware
//
#include <hip/hip_runtime.h>

// ---------------------------------------------------------------------------
// LSTM layer for MI455X (gfx1250, wave32, WMMA + async global->LDS copies).
// B=32, T=512, D=1024, H=1024, 4H=4096.
// ---------------------------------------------------------------------------

#define B_   32
#define T_   512
#define D_   1024
#define H_   1024
#define G_   4096          // 4*H
#define M1_  (B_*T_)       // 16384 rows of x_gates GEMM

typedef __bf16 bf16_t;
typedef __attribute__((ext_vector_type(16))) bf16_t       v16bf;
typedef __attribute__((ext_vector_type(8)))  float        v8f;
typedef __attribute__((ext_vector_type(4)))  unsigned int v4u;

union FragBF {
    v16bf v;
    v4u   q[2];
    unsigned short u[16];
};

// f32 -> bf16 (round-to-nearest-even), raw ushort
__device__ __forceinline__ unsigned short f2bf(float f) {
    unsigned u = __float_as_uint(f);
    unsigned r = u + 0x7FFFu + ((u >> 16) & 1u);
    return (unsigned short)(r >> 16);
}

// ---- fragment-major swizzle slots (ISA 7.12.2 wave32 layouts) -------------
// A subtile = 16x32 (MxK): lane = (m&15) + 16*((k>>3)&1); idx = (k&7)|((k>>4)<<3)
__device__ __forceinline__ int a_slot(int m, int k) {
    int lane = (m & 15) + (((k >> 3) & 1) << 4);
    int idx  = (k & 7) | ((k >> 4) << 3);
    return lane * 16 + idx;
}
// B subtile = 32x16 (KxN): lane = (n&15) + 16*((k>>3)&1); idx = (k&7)|((k>>4)<<3)
__device__ __forceinline__ int b_slot(int k, int n) {
    int lane = (n & 15) + (((k >> 3) & 1) << 4);
    int idx  = (k & 7) | ((k >> 4) << 3);
    return lane * 16 + idx;
}

// fragment load: 32 contiguous bytes per lane -> 2x ds_load_b128
__device__ __forceinline__ FragBF load_frag(const unsigned short* sub, int lane) {
    FragBF f;
    const v4u* p = (const v4u*)(sub + (size_t)lane * 16);
    f.q[0] = p[0];
    f.q[1] = p[1];
    return f;
}

__device__ __forceinline__ v8f wmma_bf16(const FragBF& a, const FragBF& b, v8f c) {
    return __builtin_amdgcn_wmma_f32_16x16x32_bf16(
        false, a.v, false, b.v, (short)0, c, false, false);
}

// ---- async global->LDS copies (gfx1250, tracked by ASYNCcnt) --------------
__device__ __forceinline__ unsigned lds_off(const void* p) {
    return (unsigned)(unsigned long long)p;   // low 32 bits of generic shared ptr
}
__device__ __forceinline__ void async_b128(unsigned dst_lds, const void* src) {
    asm volatile("global_load_async_to_lds_b128 %0, %1, off"
                 :: "v"(dst_lds), "v"((unsigned long long)src) : "memory");
}
__device__ __forceinline__ void async_b64(unsigned dst_lds, const void* src) {
    asm volatile("global_load_async_to_lds_b64 %0, %1, off"
                 :: "v"(dst_lds), "v"((unsigned long long)src) : "memory");
}
template <int N>
__device__ __forceinline__ void wait_async() {
    asm volatile("s_wait_asynccnt %0" :: "i"(N) : "memory");
}

__device__ __forceinline__ float sigmoidf_(float x) {
    return 1.0f / (1.0f + __expf(-x));
}

// ---------------------------------------------------------------------------
// Kernel 0: init workspace (h_bf16=0, c_buf=0, barrier words=0)
// ---------------------------------------------------------------------------
__global__ void k_init(unsigned short* hbf, float* cbuf, unsigned* bar) {
    int u = blockIdx.x * blockDim.x + threadIdx.x;
    if (u < B_ * H_) { hbf[u] = 0; cbuf[u] = 0.0f; }
    if (u == 0) { bar[0] = 0u; bar[1] = 0u; }
}

// ---------------------------------------------------------------------------
// Kernel 1: pack a K=1024 x N=4096 f32 matrix into bf16 fragment-major tiles:
//   dst[((k>>5)*256 + (c>>4))*512 + b_slot(k&31, c&15)]
// ---------------------------------------------------------------------------
__global__ void k_pack_b(const float* __restrict__ src, unsigned short* __restrict__ dst) {
    int i = blockIdx.x * blockDim.x + threadIdx.x;
    if (i >= H_ * G_) return;
    int k = i >> 12;            // row (K dim)
    int c = i & (G_ - 1);       // col (N dim)
    size_t blk = ((size_t)(k >> 5) * (G_ / 16) + (c >> 4)) * 512;
    dst[blk + b_slot(k & 31, c & 15)] = f2bf(src[i]);
}

// ---------------------------------------------------------------------------
// Kernel 2: pack input [16384 x 1024] f32 into bf16 A-fragment tiles:
//   dst[((r>>4)*32 + (k>>5))*512 + a_slot(r&15, k&31)]
// ---------------------------------------------------------------------------
__global__ void k_pack_a(const float* __restrict__ src, unsigned short* __restrict__ dst) {
    size_t i = (size_t)blockIdx.x * blockDim.x + threadIdx.x;
    if (i >= (size_t)M1_ * D_) return;
    int r = (int)(i >> 10);
    int k = (int)(i & (D_ - 1));
    size_t blk = ((size_t)(r >> 4) * (D_ / 32) + (k >> 5)) * 512;
    dst[blk + a_slot(r & 15, k & 31)] = f2bf(src[i]);
}

// ---------------------------------------------------------------------------
// Kernel 3: x_gates[16384,4096] = input @ W_x + bx + bh
// 128 threads (4 waves), 64x64 tile, double-buffered async tile staging.
// ---------------------------------------------------------------------------
__global__ void __launch_bounds__(128)
k_xgates(const unsigned short* __restrict__ isw,   // packed input A tiles
         const unsigned short* __restrict__ xsw,   // packed W_x B tiles
         const float* __restrict__ bx, const float* __restrict__ bh,
         float* __restrict__ xg) {
    __shared__ alignas(16) unsigned short Ab[2][4 * 512];   // 4 A subtiles / buf
    __shared__ alignas(16) unsigned short Bb[2][4 * 512];   // 4 B subtiles / buf

    const int tid  = threadIdx.x;
    const int wave = tid >> 5;
    const int lane = tid & 31;
    const int rb0  = blockIdx.y * 4;   // first 16-row block (gM0/16)
    const int nt0  = blockIdx.x * 4;   // first 16-col subtile (gN0/16)

    const unsigned la[2] = { lds_off(&Ab[0][0]), lds_off(&Ab[1][0]) };
    const unsigned lb[2] = { lds_off(&Bb[0][0]), lds_off(&Bb[1][0]) };

    v8f acc[4];
#pragma unroll
    for (int i = 0; i < 4; ++i) acc[i] = (v8f){};

    // issue one buffer: A = 4x1KB blocks, B = 4KB contiguous; 4 asyncs/wave
    auto issue = [&](int kc, int buf) {
#pragma unroll
        for (int it = 0; it < 2; ++it) {
            int q  = tid * 2 + it;              // 0..255 chunks of 16B
            int rb = q >> 6, w = q & 63;
            async_b128(la[buf] + (unsigned)(rb * 512 + w * 8) * 2,
                       isw + ((size_t)(rb0 + rb) * 32 + kc) * 512 + w * 8);
            async_b128(lb[buf] + (unsigned)q * 16,
                       xsw + ((size_t)kc * 256 + nt0) * 512 + q * 8);
        }
    };

    issue(0, 0);
    for (int kc = 0; kc < 32; ++kc) {
        const int cur = kc & 1;
        if (kc + 1 < 32) { issue(kc + 1, cur ^ 1); wait_async<4>(); }
        else             { wait_async<0>(); }
        __syncthreads();

        FragBF a = load_frag(&Ab[cur][wave * 512], lane);
#pragma unroll
        for (int nt = 0; nt < 4; ++nt) {
            FragBF b = load_frag(&Bb[cur][nt * 512], lane);
            acc[nt] = wmma_bf16(a, b, acc[nt]);
        }
        __syncthreads();
    }

    // epilogue: add bias, store f32
    const int n    = lane & 15;
    const int mofs = (lane >= 16) ? 8 : 0;
#pragma unroll
    for (int nt = 0; nt < 4; ++nt) {
        int col = nt0 * 16 + nt * 16 + n;
        float bias = bx[col] + bh[col];
#pragma unroll
        for (int v = 0; v < 8; ++v) {
            int row = rb0 * 16 + wave * 16 + v + mofs;
            xg[(size_t)row * G_ + col] = acc[nt][v] + bias;
        }
    }
}

// ---------------------------------------------------------------------------
// Global barrier (arrival counter + generation in workspace)
// ---------------------------------------------------------------------------
__device__ __forceinline__ void global_barrier(unsigned* bar, unsigned nwg) {
    __syncthreads();
    if (threadIdx.x == 0) {
        __threadfence();
        volatile unsigned* vcnt = bar;
        volatile unsigned* vgen = bar + 1;
        unsigned g = *vgen;
        if (atomicAdd(bar, 1u) == nwg - 1u) {
            *vcnt = 0u;
            __threadfence();
            atomicAdd(bar + 1, 1u);
        } else {
            while (*vgen == g) { __builtin_amdgcn_s_sleep(2); }
        }
        __threadfence();
    }
    __syncthreads();
}

// ---------------------------------------------------------------------------
// Kernel 4: persistent recurrent scan. 32 WGs x 256 threads (8 waves).
// Each WG owns a 128-column slab of the 4096 gate dims.
// h is republished each step in bf16 A-fragment-major layout:
//   hbf[(j>>5)*1024 + (b>>4)*512 + a_slot(b&15, j&31)]
// ---------------------------------------------------------------------------
__global__ void __launch_bounds__(256)
k_recurrent(const unsigned short* __restrict__ whsw,  // packed W_h B tiles
            const float* __restrict__ xg,
            unsigned short* __restrict__ hbf,         // packed h A tiles
            float* __restrict__ gbuf,
            float* __restrict__ cbuf,
            unsigned* bar,
            float* __restrict__ out_h,   // raw h -> d_out region 0
            float* __restrict__ out_c) { // raw c -> d_out region 2
    __shared__ alignas(16) unsigned short Wb[2][8 * 512];   // 8 B subtiles / buf
    __shared__ alignas(16) unsigned short Hb[2][2 * 512];   // 2 A subtiles / buf

    const int tid   = threadIdx.x;
    const int wave  = tid >> 5;
    const int lane  = tid & 31;
    const int wgN0  = blockIdx.x * 128;
    const int ntb   = wgN0 >> 4;               // first B subtile of slab
    const int gtid  = blockIdx.x * 256 + tid;
    const unsigned NWG = 32;

    const unsigned lw[2] = { lds_off(&Wb[0][0]), lds_off(&Wb[1][0]) };
    const unsigned lh[2] = { lds_off(&Hb[0][0]), lds_off(&Hb[1][0]) };

    // issue one K-chunk: W = 8KB contiguous, H = 2KB contiguous; 3 asyncs/wave
    auto issue = [&](int kc, int buf) {
        const unsigned short* wsrc = whsw + ((size_t)kc * 256 + ntb) * 512;
#pragma unroll
        for (int it = 0; it < 2; ++it) {
            int q = tid * 2 + it;               // 0..511 chunks of 16B
            async_b128(lw[buf] + (unsigned)q * 16, wsrc + q * 8);
        }
        async_b64(lh[buf] + (unsigned)tid * 8, hbf + (size_t)kc * 1024 + tid * 4);
    };

    for (int t = 0; t < T_; ++t) {
        // ---- GEMM phase: g[0..31][wgN0..wgN0+128) ----
        v8f acc0 = (v8f){};
        v8f acc1 = (v8f){};
        issue(0, 0);
        for (int kc = 0; kc < 32; ++kc) {
            const int cur = kc & 1;
            if (kc + 1 < 32) { issue(kc + 1, cur ^ 1); wait_async<3>(); }
            else             { wait_async<0>(); }
            __syncthreads();

            FragBF b  = load_frag(&Wb[cur][wave * 512], lane);
            FragBF a0 = load_frag(&Hb[cur][0],   lane);
            FragBF a1 = load_frag(&Hb[cur][512], lane);
            acc0 = wmma_bf16(a0, b, acc0);
            acc1 = wmma_bf16(a1, b, acc1);
            __syncthreads();
        }
        // scatter g to g_buf[32][4096]
        {
            const int n    = lane & 15;
            const int colg = wgN0 + wave * 16 + n;
            const int mofs = (lane >= 16) ? 8 : 0;
#pragma unroll
            for (int v = 0; v < 8; ++v) {
                gbuf[(size_t)(v + mofs)      * G_ + colg] = acc0[v];
                gbuf[(size_t)(v + mofs + 16) * G_ + colg] = acc1[v];
            }
        }
        __threadfence();
        global_barrier(bar, NWG);

        // ---- elementwise gate phase: 32*1024 units over 8192 threads ----
#pragma unroll
        for (int rep = 0; rep < 4; ++rep) {
            int u  = gtid + rep * 8192;          // [0, 32768)
            int b_ = u >> 10;
            int j  = u & (H_ - 1);
            size_t xrow = ((size_t)b_ * T_ + t) * G_;
            size_t grow = (size_t)b_ * G_;
            float gi = xg[xrow + j]          + gbuf[grow + j];
            float gf = xg[xrow + H_ + j]     + gbuf[grow + H_ + j];
            float gg = xg[xrow + 2*H_ + j]   + gbuf[grow + 2*H_ + j];
            float go = xg[xrow + 3*H_ + j]   + gbuf[grow + 3*H_ + j];
            float i_ = sigmoidf_(gi);
            float f_ = sigmoidf_(gf);
            float g_ = tanhf(gg);
            float o_ = sigmoidf_(go);
            float c  = f_ * cbuf[u] + i_ * g_;
            float h  = o_ * tanhf(c);
            cbuf[u] = c;
            // republish h in A-fragment-major bf16 layout for next step
            hbf[(size_t)(j >> 5) * 1024 + (size_t)(b_ >> 4) * 512
                + a_slot(b_ & 15, j & 31)] = f2bf(h);
            size_t orow = ((size_t)b_ * T_ + t) * H_ + j;
            out_h[orow] = h;   // raw (mask applied later)
            out_c[orow] = c;
        }
        __threadfence();
        global_barrier(bar, NWG);
    }
}

// ---------------------------------------------------------------------------
// Kernel 5: cls_token gather (BEFORE masking, matching reference order)
// ---------------------------------------------------------------------------
__global__ void k_cls(const float* __restrict__ out_h_raw,
                      const int* __restrict__ nzi,
                      float* __restrict__ cls) {
    int u = blockIdx.x * blockDim.x + threadIdx.x;
    if (u >= B_ * H_) return;
    int b_ = u >> 10;
    int j  = u & (H_ - 1);
    int t  = nzi[b_];
    cls[(size_t)b_ * H_ + j] = out_h_raw[((size_t)b_ * T_ + t) * H_ + j];
}

// ---------------------------------------------------------------------------
// Kernel 6: mask hts in-place, duplicate to region1, mask cts.
// ---------------------------------------------------------------------------
__global__ void k_mask(float* __restrict__ out0, float* __restrict__ out1,
                       float* __restrict__ out2, const float* __restrict__ mask) {
    size_t gid = (size_t)blockIdx.x * blockDim.x + threadIdx.x;
    if (gid >= (size_t)B_ * T_ * H_) return;
    int t  = (int)((gid >> 10) & (T_ - 1));
    int b_ = (int)(gid >> 19);
    float m = mask[b_ * T_ + t];
    float h = out0[gid] * m;
    out0[gid] = h;
    out1[gid] = h;
    out2[gid] = out2[gid] * m;
}

// ---------------------------------------------------------------------------
// Host-side launch
// ---------------------------------------------------------------------------
extern "C" void kernel_launch(void* const* d_in, const int* in_sizes, int n_in,
                              void* d_out, int out_size, void* d_ws, size_t ws_size,
                              hipStream_t stream) {
    const float* inp  = (const float*)d_in[0];   // [B,T,D]
    const float* mask = (const float*)d_in[1];   // [B,T]
    const int*   nzi  = (const int*)  d_in[2];   // [B,1]
    const float* wx   = (const float*)d_in[3];   // [D,4H]
    const float* wh   = (const float*)d_in[4];   // [H,4H]
    const float* bx   = (const float*)d_in[5];   // [4H]
    const float* bh   = (const float*)d_in[6];   // [4H]

    float* out  = (float*)d_out;
    float* out0 = out;                                   // hts
    float* out1 = out + (size_t)B_ * T_ * H_;            // hts copy
    float* out2 = out + 2 * (size_t)B_ * T_ * H_;        // cts
    float* cls  = out + 3 * (size_t)B_ * T_ * H_;        // [B,H]

    // workspace carve-up
    char* base = (char*)d_ws;
    size_t off = 0;
    float* xg = (float*)(base + off);                     off += (size_t)M1_ * G_ * sizeof(float);           // 256 MB
    unsigned short* isw  = (unsigned short*)(base + off); off += (size_t)M1_ * D_ * sizeof(unsigned short);  // 32 MB
    unsigned short* xsw  = (unsigned short*)(base + off); off += (size_t)D_ * G_ * sizeof(unsigned short);   // 8 MB
    unsigned short* whsw = (unsigned short*)(base + off); off += (size_t)H_ * G_ * sizeof(unsigned short);   // 8 MB
    float* gbuf = (float*)(base + off);                   off += (size_t)B_ * G_ * sizeof(float);            // 512 KB
    unsigned short* hbf = (unsigned short*)(base + off);  off += (size_t)B_ * H_ * sizeof(unsigned short);   // 64 KB
    float* cbuf = (float*)(base + off);                   off += (size_t)B_ * H_ * sizeof(float);            // 128 KB
    unsigned* bar = (unsigned*)(base + off);              off += 2 * sizeof(unsigned);

    // 0) init state
    k_init<<<dim3((B_ * H_ + 255) / 256), dim3(256), 0, stream>>>(hbf, cbuf, bar);

    // 1) pack weights + input into bf16 fragment-major tiles
    k_pack_b<<<dim3((H_ * G_ + 255) / 256), dim3(256), 0, stream>>>(wx, xsw);
    k_pack_b<<<dim3((H_ * G_ + 255) / 256), dim3(256), 0, stream>>>(wh, whsw);
    {
        size_t n = (size_t)M1_ * D_;
        k_pack_a<<<dim3((unsigned)((n + 255) / 256)), dim3(256), 0, stream>>>(inp, isw);
    }

    // 2) x_gates GEMM: grid (N/64, M/64) = (64, 256), 128 threads
    k_xgates<<<dim3(G_ / 64, M1_ / 64), dim3(128), 0, stream>>>(isw, xsw, bx, bh, xg);

    // 3) persistent recurrent scan: 32 WGs x 256
    k_recurrent<<<dim3(32), dim3(256), 0, stream>>>(whsw, xg, hbf, gbuf, cbuf, bar, out0, out2);

    // 4) cls gather BEFORE masking
    k_cls<<<dim3((B_ * H_ + 255) / 256), dim3(256), 0, stream>>>(out0, nzi, cls);

    // 5) mask + duplicate
    size_t nelem = (size_t)B_ * T_ * H_;
    k_mask<<<dim3((unsigned)((nelem + 255) / 256)), dim3(256), 0, stream>>>(out0, out1, out2, mask);
}
